// SimplifiedEncoder_22771916604100
// MI455X (gfx1250) — matrix operand order repaired and checked by reading, not compiled, not run on hardware
//
#include <hip/hip_runtime.h>
#include <hip/hip_bf16.h>

// ---------------------------------------------------------------------------
// Problem constants (from reference)
// ---------------------------------------------------------------------------
#define SEQ   512
#define BATCH 256
#define EH    256          // E == H == 256
#define G4    1024         // 4*H
#define NWG   32
#define NWAVE 8            // waves per workgroup (wave32)

typedef __attribute__((ext_vector_type(16))) __bf16 v16bf;
typedef __attribute__((ext_vector_type(8)))  float  v8f;

// packed weight matrix: [kTile(8)][nTileG(64)][lane(32)][half(16)]  (bf16)
// Bih and Bhh are contiguous, so a layer is one [kTile(16)][...] matrix.
#define PACK_HALFS (8 * 64 * 32 * 16)     // 262144 halfs = 512 KB / matrix

// LDS A-block: 16 rows x 512 bf16 (input | recurrent), pitch padded to 520
// halfs so the 16 row-lanes of an A-tile hit distinct banks (260 dw % 64 = 4).
#define ROWP 520

// workspace layout (bytes)
#define OFF_BIH0   ((size_t)0)
#define OFF_BHH0   (OFF_BIH0 + (size_t)PACK_HALFS * 2)   // must follow BIH0
#define OFF_BIH1   (OFF_BHH0 + (size_t)PACK_HALFS * 2)
#define OFF_BHH1   (OFF_BIH1 + (size_t)PACK_HALFS * 2)   // must follow BIH1
#define OFF_BIAS0  (OFF_BHH1 + (size_t)PACK_HALFS * 2)
#define OFF_BIAS1  (OFF_BIAS0 + (size_t)G4 * 4)
#define OFF_IBUF   (OFF_BIAS1 + (size_t)G4 * 4)                 // bf16 B*H  (layer0 raw h)
#define OFF_HBUF0  (OFF_IBUF  + (size_t)BATCH * EH * 2)         // bf16 B*H  (carry h)
#define OFF_HBUF1  (OFF_HBUF0 + (size_t)BATCH * EH * 2)         // bf16 B*H  (masked h after L0)
#define OFF_CBUF   (OFF_HBUF1 + (size_t)BATCH * EH * 2)         // f32  B*H  (cell state)
#define OFF_CTRL   (OFF_CBUF  + (size_t)BATCH * EH * 4)         // 2x u32 barrier state
#define WS_TOTAL   (OFF_CTRL + 64)

// ---------------------------------------------------------------------------
// Weight packing:  gates = inp @ W^T  ->  B-tile element (k, n) = W[n][k].
// WMMA bf16 B operand (32x16), contiguous K order used here (A matches): lanes 0-15 hold K=0..15 (N=lane),
// lanes 16-31 hold K=16..31; VGPR v halves = K {2v, 2v+1}.  Each lane's 16
// contiguous packed halfs = 16 contiguous K of one weight row -> two
// global_load_b128 per tile at run time.
// ---------------------------------------------------------------------------
__global__ void pack_w_kernel(const float* __restrict__ W, __bf16* __restrict__ P) {
    int idx  = blockIdx.x * 256 + threadIdx.x;        // 0 .. PACK_HALFS-1
    int j    = idx & 15;
    int lane = (idx >> 4) & 31;
    int ntg  = (idx >> 9) & 63;
    int kt   = idx >> 15;                             // 0..7
    int n    = ntg * 16 + (lane & 15);
    int k    = kt * 32 + j + ((lane >> 4) << 4);
    P[idx] = (__bf16)W[n * EH + k];
}

__global__ void bias_sum_kernel(const float* __restrict__ a0, const float* __restrict__ b0,
                                const float* __restrict__ a1, const float* __restrict__ b1,
                                float* __restrict__ s0, float* __restrict__ s1) {
    int i = blockIdx.x * 256 + threadIdx.x;           // 0..2047
    if (i < G4) s0[i] = a0[i] + b0[i];
    else        s1[i - G4] = a1[i - G4] + b1[i - G4];
}

// ---------------------------------------------------------------------------
// Operand helpers
// ---------------------------------------------------------------------------
union VB { v16bf v; uint4 q[2]; __bf16 h[16]; };

// Cooperative: gather 16 embedding rows (fp32 -> bf16) into LDS cols [0,256).
__device__ inline void stage_emb_block(const float* __restrict__ emb,
                                       const int* __restrict__ xt,
                                       __bf16* __restrict__ s, int b0) {
    const int tid = threadIdx.x;
    const int row = tid >> 4;            // 0..15
    const int col = (tid & 15) << 4;     // 0,16,..,240
    const int tok = xt[b0 + row];
    const float4* src = (const float4*)(emb + (size_t)tok * EH + col);
    float4 f0 = src[0], f1 = src[1], f2 = src[2], f3 = src[3];
    VB u;
    u.h[0] = (__bf16)f0.x;  u.h[1] = (__bf16)f0.y;  u.h[2] = (__bf16)f0.z;  u.h[3] = (__bf16)f0.w;
    u.h[4] = (__bf16)f1.x;  u.h[5] = (__bf16)f1.y;  u.h[6] = (__bf16)f1.z;  u.h[7] = (__bf16)f1.w;
    u.h[8] = (__bf16)f2.x;  u.h[9] = (__bf16)f2.y;  u.h[10] = (__bf16)f2.z; u.h[11] = (__bf16)f2.w;
    u.h[12] = (__bf16)f3.x; u.h[13] = (__bf16)f3.y; u.h[14] = (__bf16)f3.z; u.h[15] = (__bf16)f3.w;
    uint4* dst = (uint4*)(s + row * ROWP + col);
    dst[0] = u.q[0];
    dst[1] = u.q[1];
}

// Cooperative: copy a 16x256 bf16 block (global) into LDS cols [cbase, cbase+256).
__device__ inline void stage_bf16_block(const __bf16* __restrict__ g,
                                        __bf16* __restrict__ s, int b0, int cbase) {
    const int tid = threadIdx.x;
    const int row = tid >> 4;
    const int col = (tid & 15) << 4;
    const uint4* src = (const uint4*)(g + (b0 + row) * EH + col);
    uint4* dst = (uint4*)(s + row * ROWP + cbase + col);
    dst[0] = src[0];
    dst[1] = src[1];
}

// A operand (16x32 bf16) from LDS: lane L<16 holds row L,
// K 0..15; lane L>=16 holds row L-16, K 16..31 (matches packed B).
__device__ inline v16bf load_a_lds(const __bf16* __restrict__ s, int k0, int lane) {
    const __bf16* p = s + (lane & 15) * ROWP + k0 + ((lane >> 4) << 4);
    VB u;
    u.q[0] = *(const uint4*)(p);        // K 0..7 of half (8 halfs = 16B)  -> ds_load_b128
    u.q[1] = *(const uint4*)(p + 8);    // K 8..15 of half
    return u.v;
}

__device__ inline v16bf load_b_tile(const __bf16* __restrict__ W, int kt, int ntg, int lane) {
    const __bf16* p = W + ((((size_t)(kt * 64 + ntg)) * 32 + lane) << 4);
    VB u;
    u.q[0] = *(const uint4*)(p);
    u.q[1] = *(const uint4*)(p + 8);
    return u.v;
}

struct Acc4 { v8f i, f, g, o; };

__device__ inline void gemm4(Acc4& acc, v16bf a, const __bf16* __restrict__ W,
                             int kt, int nt, int lane) {
    // L2-resident weight stream; hint the next K-tile (wrap to stay in range).
    __builtin_prefetch(W + ((((size_t)((((kt + 1) & 15)) * 64 + nt)) * 32 + lane) << 4), 0, 1);
    acc.i = __builtin_amdgcn_wmma_f32_16x16x32_bf16(false, a, false, load_b_tile(W, kt, nt,      lane), (short)0, acc.i, false, false);
    acc.f = __builtin_amdgcn_wmma_f32_16x16x32_bf16(false, a, false, load_b_tile(W, kt, nt + 16, lane), (short)0, acc.f, false, false);
    acc.g = __builtin_amdgcn_wmma_f32_16x16x32_bf16(false, a, false, load_b_tile(W, kt, nt + 32, lane), (short)0, acc.g, false, false);
    acc.o = __builtin_amdgcn_wmma_f32_16x16x32_bf16(false, a, false, load_b_tile(W, kt, nt + 48, lane), (short)0, acc.o, false, false);
}

__device__ inline float sigf(float x) { return 1.0f / (1.0f + __expf(-x)); }

// ---------------------------------------------------------------------------
// Grid-wide barrier (persistent cooperative kernel)
// ---------------------------------------------------------------------------
__device__ inline void grid_sync(unsigned* cnt, unsigned* gen, unsigned phase) {
    __syncthreads();
    if (threadIdx.x == 0) {
        __threadfence();
        unsigned prev = atomicAdd(cnt, 1u);
        if (prev == NWG - 1) {
            *cnt = 0u;                       // all arrived; safe to reset
            __threadfence();
            atomicAdd(gen, 1u);
        } else {
            while (__hip_atomic_load(gen, __ATOMIC_ACQUIRE, __HIP_MEMORY_SCOPE_AGENT) <= phase) {
                __builtin_amdgcn_s_sleep(2);
            }
        }
    }
    __syncthreads();
    __threadfence();
}

// ---------------------------------------------------------------------------
// GEMM phase: 4 gate accumulators over fused K = E (input) + H (recurrent).
// Bcat = [Bih | Bhh] packed contiguously -> single 16-iteration K loop.
// unroll 2 lets iteration k+1 loads issue under iteration k WMMAs without
// blowing the VGPR budget (acc 32 + 2 A tiles + B stream ~ 150 VGPRs).
// ---------------------------------------------------------------------------
__device__ inline Acc4 run_layer(const __bf16* __restrict__ sA,
                                 const __bf16* __restrict__ Bcat,
                                 int nt, int lane) {
    Acc4 acc;
    acc.i = {}; acc.f = {}; acc.g = {}; acc.o = {};
#pragma unroll 2
    for (int kt = 0; kt < 16; ++kt) {
        v16bf a = load_a_lds(sA, kt * 32, lane);
        gemm4(acc, a, Bcat, kt, nt, lane);
    }
    return acc;
}

// D-tile layout (ISA §7.12.2, 32-bit C/D 16x16): lane l, VGPR r -> element
// (M = r + 8*(l>=16), N = l%16).  Each (b,n) cell has a unique owner wave.
template <bool WRITE_RAW>
__device__ inline void epilogue(const Acc4& acc, const float* __restrict__ bias,
                                const int* __restrict__ xt,
                                float* __restrict__ cbuf,
                                const __bf16* __restrict__ h_fallback,
                                __bf16* __restrict__ h_masked_out,
                                __bf16* __restrict__ h_raw_out,
                                int b0, int nt, int lane) {
    const int n  = nt * 16 + (lane & 15);
    const int hi = lane >> 4;
    const float bi  = bias[n];
    const float bf_ = bias[256 + n];
    const float bg  = bias[512 + n];
    const float bo  = bias[768 + n];
#pragma unroll
    for (int r = 0; r < 8; ++r) {
        const int b   = b0 + r + 8 * hi;
        const int idx = b * EH + n;
        float iv = sigf(acc.i[r] + bi);
        float fv = sigf(acc.f[r] + bf_);
        float gv = tanhf(acc.g[r] + bg);
        float ov = sigf(acc.o[r] + bo);
        float cold = cbuf[idx];
        float cnew = fv * cold + iv * gv;
        float hnew = ov * tanhf(cnew);
        bool  m    = (xt[b] != 0);
        if (WRITE_RAW) h_raw_out[idx] = (__bf16)hnew;       // unmasked, feeds layer 1
        float hfb = (float)h_fallback[idx];
        h_masked_out[idx] = (__bf16)(m ? hnew : hfb);
        cbuf[idx] = m ? cnew : cold;
    }
}

// ---------------------------------------------------------------------------
// Persistent LSTM scan: 32 WGs x 8 waves = 256 waves; all 8 waves of a WG
// share one batch tile (b0), each owns a 16-wide hidden-column tile.
// ---------------------------------------------------------------------------
__global__ void __launch_bounds__(256)
lstm_persistent_kernel(const int* __restrict__ x,        // (S, B)
                       const float* __restrict__ emb,    // (NUM_EMB, E)
                       const __bf16* __restrict__ Bcat0, // [Bih0 | Bhh0]
                       const __bf16* __restrict__ Bcat1, // [Bih1 | Bhh1]
                       const float* __restrict__ bias0, const float* __restrict__ bias1,
                       __bf16* __restrict__ ibuf,        // raw h from layer 0
                       __bf16* __restrict__ hbuf0,       // carry h (in/out per step)
                       __bf16* __restrict__ hbuf1,       // masked h after layer 0
                       float* __restrict__ cbuf,         // cell state (f32)
                       unsigned* __restrict__ ctrl) {
    __shared__ __bf16 sA[16 * ROWP];     // fused A block: [input | recurrent]

    const int lane = threadIdx.x & 31;
    const int wid  = blockIdx.x * NWAVE + (threadIdx.x >> 5);  // 0..255
    const int b0   = (wid >> 4) << 4;                          // batch tile (same for all waves of WG)
    const int nt   = wid & 15;                                 // hidden-col tile
    unsigned* cnt = ctrl;
    unsigned* gen = ctrl + 1;

#pragma unroll 1
    for (unsigned t = 0; t < SEQ; ++t) {
        const int* xt = x + (size_t)t * BATCH;

        // ---- layer 0: gates = emb_t @ Wih0^T + h @ Whh0^T ------------------
        stage_emb_block(emb, xt, sA, b0);
        stage_bf16_block(hbuf0, sA, b0, 256);
        __syncthreads();
        {
            Acc4 acc = run_layer(sA, Bcat0, nt, lane);
            epilogue<true>(acc, bias0, xt, cbuf, hbuf0, hbuf1, ibuf, b0, nt, lane);
        }
        grid_sync(cnt, gen, 2 * t);        // hbuf1 / ibuf ready device-wide

        // ---- layer 1: gates = h_raw0 @ Wih1^T + h_masked @ Whh1^T ----------
        stage_bf16_block(ibuf, sA, b0, 0);
        stage_bf16_block(hbuf1, sA, b0, 256);
        __syncthreads();
        {
            Acc4 acc = run_layer(sA, Bcat1, nt, lane);
            epilogue<false>(acc, bias1, xt, cbuf, hbuf1, hbuf0, nullptr, b0, nt, lane);
        }
        grid_sync(cnt, gen, 2 * t + 1);    // hbuf0 carry ready for next step
    }
}

// Output: (h_stack(2,B,H), c_stack(2,B,H)) flat, both layer copies identical.
__global__ void writeout_kernel(const __bf16* __restrict__ h, const float* __restrict__ c,
                                float* __restrict__ out) {
    int i = blockIdx.x * 256 + threadIdx.x;   // 0 .. B*H-1
    float hv = (float)h[i];
    float cv = c[i];
    out[i]                  = hv;
    out[BATCH * EH + i]     = hv;
    out[2 * BATCH * EH + i] = cv;
    out[3 * BATCH * EH + i] = cv;
}

// ---------------------------------------------------------------------------
extern "C" void kernel_launch(void* const* d_in, const int* in_sizes, int n_in,
                              void* d_out, int out_size, void* d_ws, size_t ws_size,
                              hipStream_t stream) {
    (void)in_sizes; (void)n_in; (void)out_size;
    const int*   x    = (const int*)d_in[0];
    const float* emb  = (const float*)d_in[1];
    const float* Wih0 = (const float*)d_in[2];
    const float* Whh0 = (const float*)d_in[3];
    const float* bih0 = (const float*)d_in[4];
    const float* bhh0 = (const float*)d_in[5];
    const float* Wih1 = (const float*)d_in[6];
    const float* Whh1 = (const float*)d_in[7];
    const float* bih1 = (const float*)d_in[8];
    const float* bhh1 = (const float*)d_in[9];
    float* out = (float*)d_out;

    char* ws = (char*)d_ws;
    if (ws_size < WS_TOTAL) return;     // need ~2.7 MB scratch

    __bf16* Bih0p  = (__bf16*)(ws + OFF_BIH0);
    __bf16* Bhh0p  = (__bf16*)(ws + OFF_BHH0);
    __bf16* Bih1p  = (__bf16*)(ws + OFF_BIH1);
    __bf16* Bhh1p  = (__bf16*)(ws + OFF_BHH1);
    float*  bias0  = (float*)(ws + OFF_BIAS0);
    float*  bias1  = (float*)(ws + OFF_BIAS1);
    __bf16* ibuf   = (__bf16*)(ws + OFF_IBUF);
    __bf16* hbuf0  = (__bf16*)(ws + OFF_HBUF0);
    __bf16* hbuf1  = (__bf16*)(ws + OFF_HBUF1);
    float*  cbuf   = (float*)(ws + OFF_CBUF);
    unsigned* ctrl = (unsigned*)(ws + OFF_CTRL);

    // zero h carry, cell state, barrier state (and ibuf/hbuf1, harmless)
    hipMemsetAsync(ws + OFF_IBUF, 0, WS_TOTAL - OFF_IBUF, stream);

    // one-time per call: pack weights to bf16 WMMA-B layout, fuse biases
    pack_w_kernel<<<PACK_HALFS / 256, 256, 0, stream>>>(Wih0, Bih0p);
    pack_w_kernel<<<PACK_HALFS / 256, 256, 0, stream>>>(Whh0, Bhh0p);
    pack_w_kernel<<<PACK_HALFS / 256, 256, 0, stream>>>(Wih1, Bih1p);
    pack_w_kernel<<<PACK_HALFS / 256, 256, 0, stream>>>(Whh1, Bhh1p);
    bias_sum_kernel<<<(2 * G4) / 256, 256, 0, stream>>>(bih0, bhh0, bih1, bhh1, bias0, bias1);

    // persistent cooperative scan over 512 timesteps
    lstm_persistent_kernel<<<NWG, 256, 0, stream>>>(x, emb, Bih0p, Bih1p,
                                                    bias0, bias1, ibuf, hbuf0, hbuf1,
                                                    cbuf, ctrl);

    writeout_kernel<<<(BATCH * EH) / 256, 256, 0, stream>>>(hbuf0, cbuf, out);
}
